// CKT_21320217657952
// MI455X (gfx1250) — matrix operand order repaired and checked by reading, not compiled, not run on hardware
//
#include <hip/hip_runtime.h>
#include <hip/hip_bf16.h>

// ---------------------------------------------------------------------------
// Types / WMMA helpers (gfx1250 wave32)
// ---------------------------------------------------------------------------
typedef __bf16 bf16_t;
typedef __attribute__((ext_vector_type(16))) __bf16 bfrag;   // 16x16x32 A/B frag
typedef __attribute__((ext_vector_type(8)))  float  v8f;     // 16x16 f32 C/D frag

union BFragU { bfrag f; uint4 q[2]; };

// A-matrix 16x32 bf16 fragment, source row-major [M x K], base = &A[0][k0].
// Layout: lane (m = lane&15, half = lane>>4); elem e: K = (e>=8?16:0)+half*8+(e&7)
__device__ __forceinline__ bfrag load_a_frag(const bf16_t* base, int lda, int lane) {
    int m = lane & 15, half = (lane >> 4) & 1;
    const bf16_t* p = base + (long long)m * lda + half * 8;
    BFragU u;
    u.q[0] = *(const uint4*)(p);        // elems 0..7  : K = half*8 + 0..7
    u.q[1] = *(const uint4*)(p + 16);   // elems 8..15 : K = 16 + half*8 + 0..7
    return u.f;
}

// B-matrix 32x16 bf16 fragment, source stored K-contiguous per column:
// element B[k][n] at base + n*ldn + k.  elem e: K = (e>=8?16:0)+half*8+(e&7)
__device__ __forceinline__ bfrag load_b_frag(const bf16_t* base, int ldn, int lane) {
    int n = lane & 15, half = (lane >> 4) & 1;
    const bf16_t* p = base + (long long)n * ldn + half * 8;
    BFragU u;
    u.q[0] = *(const uint4*)(p);        // elems 0..7
    u.q[1] = *(const uint4*)(p + 16);   // elems 8..15
    return u.f;
}

__device__ __forceinline__ v8f wmma_bf16(bfrag a, bfrag b, v8f c) {
    return __builtin_amdgcn_wmma_f32_16x16x32_bf16(
        /*neg_a=*/false, a, /*neg_b=*/false, b,
        /*c_mod=*/(short)0, c, /*reuse_a=*/false, /*reuse_b=*/false);
}

__device__ __forceinline__ float sigmoidf_(float x) { return 1.f / (1.f + __expf(-x)); }

#define B_  64
#define L_  1024
#define D_  256
#define C_  256
#define K_  4
#define NL_ 3
#define HD_ 1280   // 4D + C

// ---------------------------------------------------------------------------
// Weight prep: fp32 -> bf16 (W1/W2 already K-contiguous); conv transposed to
// [l][k][o][d] so the B-fragment K (=d) dimension is contiguous.
// ---------------------------------------------------------------------------
__global__ void prep_cast_kernel(const float* __restrict__ src, bf16_t* __restrict__ dst, int n) {
    int i = blockIdx.x * blockDim.x + threadIdx.x;
    if (i < n) dst[i] = (bf16_t)src[i];
}

__global__ void prep_conv_kernel(const float* __restrict__ src, bf16_t* __restrict__ dst) {
    // src: [NL][K][D][2D]   dst: [NL][K][2D][D]
    int i = blockIdx.x * blockDim.x + threadIdx.x;
    int total = NL_ * K_ * D_ * 2 * D_;
    if (i >= total) return;
    int o = i % (2 * D_);
    int d = (i / (2 * D_)) % D_;
    int lk = i / (2 * D_ * D_);          // l*K + k
    dst[((long long)lk * 2 * D_ + o) * D_ + d] = (bf16_t)src[i];
}

// ---------------------------------------------------------------------------
// Embedding gather: builds qe(bf16), LIS-transposed(bf16), and H columns
// [0..2D) = LIS, [4D..4D+C) = cqc.  HRP columns filled by attention kernel.
// ---------------------------------------------------------------------------
__global__ void embed_kernel(const int* __restrict__ qseq, const int* __restrict__ cseq,
                             const float* __restrict__ cqc, const float* __restrict__ Eq,
                             const float* __restrict__ Ec,
                             bf16_t* __restrict__ qe_bf, bf16_t* __restrict__ lis_t,
                             bf16_t* __restrict__ H) {
    int bl = blockIdx.x;                 // b*L + t
    int b = bl >> 10, t = bl & (L_ - 1);
    int d = threadIdx.x;
    float qv = Eq[(long long)qseq[bl] * D_ + d];
    float cv = Ec[(long long)cseq[bl] * D_ + d];
    qe_bf[(long long)bl * D_ + d] = (bf16_t)qv;
    bf16_t* Hrow = H + (long long)bl * HD_;
    Hrow[d]        = (bf16_t)qv;
    Hrow[D_ + d]   = (bf16_t)cv;
    Hrow[4 * D_ + d] = (bf16_t)cqc[(long long)bl * C_ + d];
    bf16_t* lt = lis_t + (long long)b * 2 * D_ * L_;
    lt[(long long)d * L_ + t]        = (bf16_t)qv;
    lt[(long long)(D_ + d) * L_ + t] = (bf16_t)cv;
}

// ---------------------------------------------------------------------------
// Attention: one block = 16 query rows of one batch.  8 waves.
// Phase A: scores via WMMA -> LDS (bf16, masked).  Block softmax matching
// reference (softmax over j<=i incl. diag, zero diag, renorm + 1e-8).
// Phase B: HRP = P @ LIS via WMMA, written into H columns [2D..4D).
// ---------------------------------------------------------------------------
__global__ __launch_bounds__(256) void attn_kernel(const bf16_t* __restrict__ qe_bf,
                                                   const bf16_t* __restrict__ lis_t,
                                                   bf16_t* __restrict__ H) {
    __shared__ __align__(16) bf16_t sc[16][L_];      // 32 KB: scores then probs
    __shared__ float red[16][16], red2[16][16];
    __shared__ float rowM[16], rowD[16];

    int blk = blockIdx.x;
    int b = blk >> 6, it = blk & 63;     // 64 row-tiles per batch
    int t0 = it * 16;
    int lane = threadIdx.x & 31, wave = threadIdx.x >> 5, half = lane >> 4;

    // zero prob buffer (tail tiles must read as 0)
    uint4 z4 = make_uint4(0u, 0u, 0u, 0u);
    for (int i = threadIdx.x; i < 16 * L_ / 8; i += 256) ((uint4*)sc)[i] = z4;
    __syncthreads();

    const bf16_t* qeB = qe_bf + (long long)b * L_ * D_;

    // -------- Phase A: masked score tiles --------
    for (int jt = wave; jt <= it; jt += 8) {
        v8f acc = {0.f, 0.f, 0.f, 0.f, 0.f, 0.f, 0.f, 0.f};
        for (int kt = 0; kt < D_ / 32; ++kt) {
            bfrag a  = load_a_frag(qeB + (long long)t0 * D_ + kt * 32, D_, lane);
            bfrag bb = load_b_frag(qeB + (long long)jt * 16 * D_ + kt * 32, D_, lane);
            acc = wmma_bf16(a, bb, acc);
        }
        int jn = jt * 16 + (lane & 15);
        for (int r = 0; r < 8; ++r) {
            int row = r + half * 8;
            float s = (jn <= t0 + row) ? acc[r] : -1e30f;
            sc[row][jn] = (bf16_t)s;
        }
    }
    __syncthreads();

    // -------- softmax stats: row = tid&15, chunk of 64 j's = tid>>4 --------
    int row = threadIdx.x & 15, chunk = threadIdx.x >> 4;
    int ig = t0 + row;                   // valid j: j <= ig (diag later removed)
    float lmax = -1e30f;
    for (int j = chunk * 64; j < chunk * 64 + 64; ++j)
        if (j <= ig) lmax = fmaxf(lmax, (float)sc[row][j]);
    red[row][chunk] = lmax;
    __syncthreads();
    if (chunk == 0) {
        float M = -1e30f;
        for (int c = 0; c < 16; ++c) M = fmaxf(M, red[row][c]);
        rowM[row] = M;
    }
    __syncthreads();
    float M = rowM[row];
    float lpart = 0.f, ldiag = 0.f;
    for (int j = chunk * 64; j < chunk * 64 + 64; ++j) {
        if (j <= ig) {
            float e = __expf((float)sc[row][j] - M);
            if (j == ig) ldiag = e; else lpart += e;
        }
    }
    red[row][chunk] = lpart; red2[row][chunk] = ldiag;
    __syncthreads();
    if (chunk == 0) {
        float Sp = 0.f, Dg = 0.f;
        for (int c = 0; c < 16; ++c) { Sp += red[row][c]; Dg += red2[row][c]; }
        rowD[row] = Sp + 1e-8f * (Sp + Dg);   // renorm denom (ref: W/(sum+1e-8))
    }
    __syncthreads();
    float denom = rowD[row];
    int jlimit = (it + 1) * 16;
    for (int j = chunk * 64; j < chunk * 64 + 64; ++j) {
        if (j >= jlimit) break;
        float p = (j < ig) ? __expf((float)sc[row][j] - M) / denom : 0.f;
        sc[row][j] = (bf16_t)p;
    }
    __syncthreads();

    // -------- Phase B: HRP = P @ LIS ; wave owns 64 of 512 output cols -----
    const bf16_t* lt = lis_t + (long long)b * 2 * D_ * L_;
    v8f acc[4];
    for (int i = 0; i < 4; ++i) acc[i] = v8f{0.f,0.f,0.f,0.f,0.f,0.f,0.f,0.f};
    int nk = (it + 2) >> 1;              // ceil((it+1)*16 / 32)
    for (int kt = 0; kt < nk; ++kt) {
        bfrag a = load_a_frag(&sc[0][0] + kt * 32, L_, lane);
        for (int nt = 0; nt < 4; ++nt) {
            int o = wave * 64 + nt * 16;
            bfrag bb = load_b_frag(lt + (long long)o * L_ + kt * 32, L_, lane);
            acc[nt] = wmma_bf16(a, bb, acc[nt]);
        }
    }
    bf16_t* Hb = H + (long long)(b * L_ + t0) * HD_ + 2 * D_;
    for (int nt = 0; nt < 4; ++nt) {
        int o = wave * 64 + nt * 16 + (lane & 15);
        for (int r = 0; r < 8; ++r)
            Hb[(long long)(r + half * 8) * HD_ + o] = (bf16_t)acc[nt][r];
    }
}

// ---------------------------------------------------------------------------
// Gated MLP: Q = (H@W1^T + b1) * sigmoid(H@W2^T + b2).  16 rows/block.
// ---------------------------------------------------------------------------
__global__ __launch_bounds__(256) void mlp_kernel(const bf16_t* __restrict__ H,
                                                  const bf16_t* __restrict__ W1t,
                                                  const bf16_t* __restrict__ W2t,
                                                  const float* __restrict__ b1,
                                                  const float* __restrict__ b2,
                                                  float* __restrict__ xf, bf16_t* __restrict__ xb) {
    __shared__ __align__(16) bf16_t Ht[16][HD_];     // 40 KB
    long long row0 = (long long)blockIdx.x * 16;
    const uint4* src = (const uint4*)(H + row0 * HD_);
    uint4* dst = (uint4*)&Ht[0][0];
    for (int i = threadIdx.x; i < 16 * HD_ / 8; i += 256) dst[i] = src[i];
    __syncthreads();

    int lane = threadIdx.x & 31, wave = threadIdx.x >> 5, half = lane >> 4;
    v8f a1[2], a2[2];
    for (int i = 0; i < 2; ++i) {
        a1[i] = v8f{0.f,0.f,0.f,0.f,0.f,0.f,0.f,0.f};
        a2[i] = v8f{0.f,0.f,0.f,0.f,0.f,0.f,0.f,0.f};
    }
    for (int kt = 0; kt < HD_ / 32; ++kt) {
        bfrag a = load_a_frag(&Ht[0][0] + kt * 32, HD_, lane);
        for (int nt = 0; nt < 2; ++nt) {
            int o = wave * 32 + nt * 16;
            bfrag w1 = load_b_frag(W1t + (long long)o * HD_ + kt * 32, HD_, lane);
            a1[nt] = wmma_bf16(a, w1, a1[nt]);
            bfrag w2 = load_b_frag(W2t + (long long)o * HD_ + kt * 32, HD_, lane);
            a2[nt] = wmma_bf16(a, w2, a2[nt]);
        }
    }
    for (int nt = 0; nt < 2; ++nt) {
        int o = wave * 32 + nt * 16 + (lane & 15);
        float bb1 = b1[o], bb2 = b2[o];
        for (int r = 0; r < 8; ++r) {
            float v1 = a1[nt][r] + bb1, v2 = a2[nt][r] + bb2;
            float q = v1 * sigmoidf_(v2);
            long long idx = (row0 + r + half * 8) * D_ + o;
            xf[idx] = q; xb[idx] = (bf16_t)q;
        }
    }
}

// ---------------------------------------------------------------------------
// Causal GLU conv layer: co[t,o] = b[o] + sum_k sum_d x[t-3+k,d] w[k,d,o]
// x_new = co[:,0:D] * sigmoid(co[:,D:2D]) + x.   19-row shifted tile in LDS.
// ---------------------------------------------------------------------------
__global__ __launch_bounds__(256) void conv_kernel(const float* __restrict__ xf_in,
                                                   const bf16_t* __restrict__ xb_in,
                                                   const bf16_t* __restrict__ wt,   // [K][2D][D]
                                                   const float* __restrict__ bias,  // [2D]
                                                   float* __restrict__ xf_out,
                                                   bf16_t* __restrict__ xb_out) {
    __shared__ __align__(16) bf16_t xs[19][D_];      // 9.5 KB (rows t0-3 .. t0+15)
    __shared__ float co[16][2 * D_];                 // 32 KB
    int blk = blockIdx.x;
    int b = blk >> 6, tt = (blk & 63) * 16;
    long long rowg0 = (long long)(b * L_ + tt);

    for (int i = threadIdx.x; i < 19 * D_; i += 256) {
        int rr = i >> 8, d = i & (D_ - 1);
        int tl = tt + rr - (K_ - 1);
        xs[rr][d] = (tl >= 0) ? xb_in[(long long)(b * L_ + tl) * D_ + d] : (bf16_t)0.f;
    }
    __syncthreads();

    int lane = threadIdx.x & 31, wave = threadIdx.x >> 5, half = lane >> 4;
    v8f acc[4];
    for (int i = 0; i < 4; ++i) acc[i] = v8f{0.f,0.f,0.f,0.f,0.f,0.f,0.f,0.f};
    for (int k = 0; k < K_; ++k) {
        for (int kt = 0; kt < D_ / 32; ++kt) {
            bfrag a = load_a_frag(&xs[k][0] + kt * 32, D_, lane);  // row m+k == t0+m-3+k
            for (int nt = 0; nt < 4; ++nt) {
                int o = wave * 64 + nt * 16;
                bfrag w = load_b_frag(wt + ((long long)k * 2 * D_ + o) * D_ + kt * 32, D_, lane);
                acc[nt] = wmma_bf16(a, w, acc[nt]);
            }
        }
    }
    for (int nt = 0; nt < 4; ++nt) {
        int o = wave * 64 + nt * 16 + (lane & 15);
        for (int r = 0; r < 8; ++r) co[r + half * 8][o] = acc[nt][r];
    }
    __syncthreads();

    for (int i = threadIdx.x; i < 16 * D_; i += 256) {
        int rr = i >> 8, d = i & (D_ - 1);
        float av = co[rr][d] + bias[d];
        float gv = co[rr][D_ + d] + bias[D_ + d];
        long long idx = (rowg0 + rr) * D_ + d;
        float v = av * sigmoidf_(gv) + xf_in[idx];
        xf_out[idx] = v; xb_out[idx] = (bf16_t)v;
    }
}

// ---------------------------------------------------------------------------
// predict[b,t] = sigmoid( dot(x[b,t,:], Eq[qseq[b,t+1],:]) ),  t in [0, L-1)
// ---------------------------------------------------------------------------
__global__ void predict_kernel(const float* __restrict__ xf, const int* __restrict__ qseq,
                               const float* __restrict__ Eq, float* __restrict__ out) {
    int idx = blockIdx.x * blockDim.x + threadIdx.x;
    if (idx >= B_ * (L_ - 1)) return;
    int b = idx / (L_ - 1), t = idx % (L_ - 1);
    const float* xr = xf + (long long)(b * L_ + t) * D_;
    const float* qr = Eq + (long long)qseq[b * L_ + t + 1] * D_;
    float s = 0.f;
    for (int d = 0; d < D_; d += 4) {
        float4 xv = *(const float4*)(xr + d);
        float4 qv = *(const float4*)(qr + d);
        s += xv.x * qv.x + xv.y * qv.y + xv.z * qv.z + xv.w * qv.w;
    }
    out[idx] = sigmoidf_(s);
}

// ---------------------------------------------------------------------------
extern "C" void kernel_launch(void* const* d_in, const int* in_sizes, int n_in,
                              void* d_out, int out_size, void* d_ws, size_t ws_size,
                              hipStream_t stream) {
    (void)in_sizes; (void)n_in; (void)out_size; (void)ws_size;
    const int*   qseq   = (const int*)d_in[0];
    const int*   cseq   = (const int*)d_in[1];
    const float* cqc    = (const float*)d_in[2];
    const float* Eq     = (const float*)d_in[3];
    const float* Ec     = (const float*)d_in[4];
    const float* W1_w   = (const float*)d_in[5];
    const float* W1_b   = (const float*)d_in[6];
    const float* W2_w   = (const float*)d_in[7];
    const float* W2_b   = (const float*)d_in[8];
    const float* conv_w = (const float*)d_in[9];
    const float* conv_b = (const float*)d_in[10];
    float* out = (float*)d_out;

    char* ws = (char*)d_ws;
    size_t off = 0;
    auto alloc = [&](size_t bytes) -> void* {
        void* p = ws + off;
        off = (off + bytes + 255) & ~(size_t)255;
        return p;
    };
    bf16_t* qe_bf  = (bf16_t*)alloc((size_t)B_ * L_ * D_ * 2);
    bf16_t* lis_t  = (bf16_t*)alloc((size_t)B_ * 2 * D_ * L_ * 2);
    bf16_t* H      = (bf16_t*)alloc((size_t)B_ * L_ * HD_ * 2);
    float*  xf0    = (float*)alloc((size_t)B_ * L_ * D_ * 4);
    float*  xf1    = (float*)alloc((size_t)B_ * L_ * D_ * 4);
    bf16_t* xb0    = (bf16_t*)alloc((size_t)B_ * L_ * D_ * 2);
    bf16_t* xb1    = (bf16_t*)alloc((size_t)B_ * L_ * D_ * 2);
    bf16_t* W1t    = (bf16_t*)alloc((size_t)D_ * HD_ * 2);
    bf16_t* W2t    = (bf16_t*)alloc((size_t)D_ * HD_ * 2);
    bf16_t* convwt = (bf16_t*)alloc((size_t)NL_ * K_ * 2 * D_ * D_ * 2);

    // weight prep
    int nW = D_ * HD_;
    prep_cast_kernel<<<(nW + 255) / 256, 256, 0, stream>>>(W1_w, W1t, nW);
    prep_cast_kernel<<<(nW + 255) / 256, 256, 0, stream>>>(W2_w, W2t, nW);
    int nC = NL_ * K_ * D_ * 2 * D_;
    prep_conv_kernel<<<(nC + 255) / 256, 256, 0, stream>>>(conv_w, convwt);

    // embeddings / LIS / H skeleton
    embed_kernel<<<B_ * L_, D_, 0, stream>>>(qseq, cseq, cqc, Eq, Ec, qe_bf, lis_t, H);

    // attention (fills HRP columns of H)
    attn_kernel<<<B_ * (L_ / 16), 256, 0, stream>>>(qe_bf, lis_t, H);

    // gated MLP -> x (ping buffer 0)
    mlp_kernel<<<B_ * L_ / 16, 256, 0, stream>>>(H, W1t, W2t, W1_b, W2_b, xf0, xb0);

    // conv stack: layer i reads buf i%2, writes buf (i+1)%2
    float*  xfin[2]  = { xf0, xf1 };
    bf16_t* xbin[2]  = { xb0, xb1 };
    for (int l = 0; l < NL_; ++l) {
        conv_kernel<<<B_ * L_ / 16, 256, 0, stream>>>(
            xfin[l & 1], xbin[l & 1],
            convwt + (size_t)l * K_ * 2 * D_ * D_,
            conv_b + (size_t)l * 2 * D_,
            xfin[(l + 1) & 1], xbin[(l + 1) & 1]);
    }
    // after 3 layers the live buffer is index 1
    int np = B_ * (L_ - 1);
    predict_kernel<<<(np + 255) / 256, 256, 0, stream>>>(xf1, qseq, Eq, out);
}
